// DistanceResonanceModel_17789754540235
// MI455X (gfx1250) — compile-verified, matrix operand order checked by reading, not run on hardware
//
#include <hip/hip_runtime.h>
#include <hip/hip_bf16.h>

// ---------------------------------------------------------------------------
// DistanceResonanceModel forward, CDNA5 (gfx1250) WMMA implementation.
// B=2048, H=256, HID=512, OUT=8, T=64, BETA=0.9, THRESH=1.0
// ---------------------------------------------------------------------------

#define B_SZ   2048
#define H_SZ   256
#define HID_SZ 512
#define OUT_SZ 8
#define T_SZ   64
#define BETA   0.9f

typedef __attribute__((ext_vector_type(16))) _Float16 v16h;
typedef __attribute__((ext_vector_type(8)))  _Float16 v8h;
typedef __attribute__((ext_vector_type(8)))  float    v8f;

// ---- WMMA helpers ---------------------------------------------------------

__device__ inline v8f wmma_f16(v16h a, v16h b, v8f c) {
  // D = A(16x32 f16) x B(32x16 f16) + C(16x16 f32)
  return __builtin_amdgcn_wmma_f32_16x16x32_f16(
      /*neg_a=*/false, a, /*neg_b=*/false, b,
      /*c_mod=*/(short)0, c, /*reuse_a=*/false, /*reuse_b=*/false);
}

// Load one 16x32 f16 A/B fragment from a row-major matrix (leading dim `ld`
// elements). base points at (row0, k0) of the tile.
// Per ISA 7.12.2: lane<16 -> row=lane,   K={0..7, 16..23}
//                 lane>=16 -> row=lane-16, K={8..15, 24..31}
__device__ inline v16h frag_ld(const _Float16* base, int ld, int lane) {
  const int r  = lane & 15;
  const int ko = (lane >> 4) << 3;           // 0 or 8
  const _Float16* p = base + r * ld + ko;
  v8h lo = *(const v8h*)(p);
  v8h hi = *(const v8h*)(p + 16);
  return __builtin_shufflevector(lo, hi, 0,1,2,3,4,5,6,7,8,9,10,11,12,13,14,15);
}

// ---- Prep kernels ---------------------------------------------------------

__global__ void cvt_f32_to_f16(const float* __restrict__ s,
                               _Float16* __restrict__ d, int n) {
  int i = blockIdx.x * blockDim.x + threadIdx.x;
  if (i < n) d[i] = (_Float16)s[i];
}

__global__ void decay_freq_kernel(const float* __restrict__ draw,
                                  const float* __restrict__ fraw,
                                  float* __restrict__ dec,
                                  float* __restrict__ frq) {
  int i = threadIdx.x;                       // H_SZ = 256 threads
  float sd = 1.0f / (1.0f + expf(-draw[i]));
  float sf = 1.0f / (1.0f + expf(-fraw[i]));
  dec[i] = 0.75f + 0.12f * sd;
  frq[i] = 0.05f + 0.20f * sf;
}

// fused_h[b][256+j]=az[b][j]; fused_h[b][512+j]=el[b][j]  (f16)
__global__ void pack_azel(const float* __restrict__ az,
                          const float* __restrict__ el,
                          _Float16* __restrict__ fused) {
  int i = blockIdx.x * blockDim.x + threadIdx.x;     // over B*H
  if (i >= B_SZ * H_SZ) return;
  int b = i >> 8, j = i & 255;
  fused[b * (3 * H_SZ) + H_SZ + j]     = (_Float16)az[i];
  fused[b * (3 * H_SZ) + 2 * H_SZ + j] = (_Float16)el[i];
}

// ---- Generic WMMA GEMM: C[M,N] = A[M,K] @ Bw[N,K]^T + bias[N]  ------------
// One wave per 16x16 output tile, 8 waves per block. Grid sized exactly.

__global__ void __launch_bounds__(256)
gemm_wmma(const _Float16* __restrict__ A, const _Float16* __restrict__ Bw,
          const float* __restrict__ bias, float* __restrict__ C,
          int M, int N, int K) {
  const int wave = threadIdx.x >> 5;
  const int lane = threadIdx.x & 31;
  const int tile = blockIdx.x * 8 + wave;
  const int ntn  = N >> 4;
  const int tm = tile / ntn, tn = tile % ntn;
  if (tm >= (M >> 4)) return;                        // wave-uniform guard

  const _Float16* Ab = A  + (tm << 4) * K;
  const _Float16* Bb = Bw + (tn << 4) * K;
  float bv = bias[(tn << 4) + (lane & 15)];
  v8f c = {bv, bv, bv, bv, bv, bv, bv, bv};

  for (int k0 = 0; k0 < K; k0 += 32) {
    v16h a = frag_ld(Ab + k0, K, lane);
    v16h b = frag_ld(Bb + k0, K, lane);
    c = wmma_f16(a, b, c);
  }
  const int mo = (lane >> 4) << 3, n = lane & 15;
  float* Cp = C + ((tm << 4) + mo) * N + (tn << 4) + n;
#pragma unroll
  for (int r = 0; r < 8; r++) Cp[r * N] = c[r];
}

// ---- Distance resonate-and-fire recurrence (fused, T=64) ------------------
// 16 batch rows per block; 256 threads = 8 waves; thread t owns column t
// of the 16xH state tile (registers). Spikes exchanged via LDS f16; the
// per-step GEMM rec = spk @ W_rec^T runs on WMMA (wave w -> N-tiles 2w,2w+1).
// The wave's W_rec slice (16 fragments, 128 VGPRs) is loop-invariant and is
// cached in registers once, eliminating all per-step weight traffic.

__global__ void __launch_bounds__(256)
dist_loop(const float* __restrict__ drive0, const float* __restrict__ decay,
          const float* __restrict__ freq, const _Float16* __restrict__ Wr,
          const float* __restrict__ b_rec, _Float16* __restrict__ fused) {
  __shared__ _Float16 spk_s[16 * H_SZ];   // 8 KB
  __shared__ float    rec_s[16 * H_SZ];   // 16 KB

  const int tid  = threadIdx.x;           // column 0..255
  const int lane = tid & 31, wave = tid >> 5;
  const int row0 = blockIdx.x << 4;

  const float dec = decay[tid], frq = freq[tid];
  float mem[16], res[16], rec[16], drv0[16], ssum[16];
#pragma unroll
  for (int r = 0; r < 16; r++) {
    mem[r] = 0.f; res[r] = 0.f; rec[r] = 0.f; ssum[r] = 0.f;
    drv0[r] = drive0[(row0 + r) * H_SZ + tid];
  }

  const int n0 = wave << 5;               // first N-tile base col
  const float bv0 = b_rec[n0 + (lane & 15)];
  const float bv1 = b_rec[n0 + 16 + (lane & 15)];
  const int mo = (lane >> 4) << 3, nn = lane & 15;

  // Register-cache the loop-invariant W_rec fragments: 2 N-tiles x 8 K-steps.
  v16h bfrag[16];
#pragma unroll
  for (int ks = 0; ks < 8; ks++) {
    bfrag[ks]     = frag_ld(Wr + n0 * H_SZ + ks * 32, H_SZ, lane);
    bfrag[8 + ks] = frag_ld(Wr + (n0 + 16) * H_SZ + ks * 32, H_SZ, lane);
  }

  for (int t = 0; t < T_SZ; t++) {
#pragma unroll
    for (int r = 0; r < 16; r++) {
      float drv = drv0[r] + rec[r];
      res[r] = dec * res[r] + frq * mem[r];
      mem[r] = dec * mem[r] + drv - res[r];
      float spk = (mem[r] > 1.0f) ? 1.0f : 0.0f;
      mem[r] -= spk;
      ssum[r] += spk;
      spk_s[r * H_SZ + tid] = (_Float16)spk;
    }
    __syncthreads();

    v8f acc0 = {bv0, bv0, bv0, bv0, bv0, bv0, bv0, bv0};
    v8f acc1 = {bv1, bv1, bv1, bv1, bv1, bv1, bv1, bv1};
#pragma unroll
    for (int ks = 0; ks < 8; ks++) {
      v16h a = frag_ld(&spk_s[ks * 32], H_SZ, lane);
      acc0 = wmma_f16(a, bfrag[ks], acc0);
      acc1 = wmma_f16(a, bfrag[8 + ks], acc1);
    }
#pragma unroll
    for (int r = 0; r < 8; r++) {
      rec_s[(r + mo) * H_SZ + n0 + nn]      = acc0[r];
      rec_s[(r + mo) * H_SZ + n0 + 16 + nn] = acc1[r];
    }
    __syncthreads();
#pragma unroll
    for (int r = 0; r < 16; r++) rec[r] = rec_s[r * H_SZ + tid];
  }

  // distance_context -> fused columns [0,256);  k/64 is exact in f16
#pragma unroll
  for (int r = 0; r < 16; r++)
    fused[(row0 + r) * (3 * H_SZ) + tid] = (_Float16)(ssum[r] * (1.0f / T_SZ));
}

// ---- Two-layer LIF recurrence (fused, T=64) + output projection -----------
// 16 rows per block; 512 threads = 16 waves; thread t owns column t of the
// 16xHID state tile. Per-step GEMM icur = fspk @ W_int^T on WMMA.
// Registers cache 4 of 16 K-steps of the wave's W_int slice (64 VGPRs),
// trimming 25% of the per-step L2 weight traffic within the VGPR budget
// of 16 waves/WGP.

__global__ void __launch_bounds__(512)
lif_loop(const float* __restrict__ fcur_g, const _Float16* __restrict__ Wi,
         const float* __restrict__ b_int, const float* __restrict__ W_out,
         const float* __restrict__ b_out, float* __restrict__ out) {
  __shared__ _Float16 fspk_s[16 * HID_SZ];  // 16 KB
  __shared__ float    icur_s[16 * HID_SZ];  // 32 KB

  const int tid  = threadIdx.x;             // column 0..511
  const int lane = tid & 31, wave = tid >> 5;
  const int row0 = blockIdx.x << 4;

  float fcur[16], fmem[16], imem[16], isum[16];
#pragma unroll
  for (int r = 0; r < 16; r++) {
    fcur[r] = fcur_g[(row0 + r) * HID_SZ + tid];
    fmem[r] = 0.f; imem[r] = 0.f; isum[r] = 0.f;
  }

  const int n0 = wave << 5;
  const float bv0 = b_int[n0 + (lane & 15)];
  const float bv1 = b_int[n0 + 16 + (lane & 15)];
  const int mo = (lane >> 4) << 3, nn = lane & 15;

  // Partial register cache of loop-invariant W_int fragments (K-steps 0..3).
  v16h bc0[4], bc1[4];
#pragma unroll
  for (int ks = 0; ks < 4; ks++) {
    bc0[ks] = frag_ld(Wi + n0 * HID_SZ + ks * 32, HID_SZ, lane);
    bc1[ks] = frag_ld(Wi + (n0 + 16) * HID_SZ + ks * 32, HID_SZ, lane);
  }

  for (int t = 0; t < T_SZ; t++) {
#pragma unroll
    for (int r = 0; r < 16; r++) {
      float rf = (fmem[r] > 1.0f) ? 1.0f : 0.0f;      // reset from old mem
      fmem[r] = BETA * fmem[r] + fcur[r] - rf;
      float fspk = (fmem[r] > 1.0f) ? 1.0f : 0.0f;
      fspk_s[r * HID_SZ + tid] = (_Float16)fspk;
    }
    __syncthreads();

    v8f acc0 = {bv0, bv0, bv0, bv0, bv0, bv0, bv0, bv0};
    v8f acc1 = {bv1, bv1, bv1, bv1, bv1, bv1, bv1, bv1};
#pragma unroll
    for (int ks = 0; ks < 16; ks++) {
      v16h a  = frag_ld(&fspk_s[ks * 32], HID_SZ, lane);
      v16h b0 = (ks < 4) ? bc0[ks & 3]
                         : frag_ld(Wi + n0 * HID_SZ + ks * 32, HID_SZ, lane);
      acc0 = wmma_f16(a, b0, acc0);
      v16h b1 = (ks < 4) ? bc1[ks & 3]
                         : frag_ld(Wi + (n0 + 16) * HID_SZ + ks * 32, HID_SZ, lane);
      acc1 = wmma_f16(a, b1, acc1);
    }
#pragma unroll
    for (int r = 0; r < 8; r++) {
      icur_s[(r + mo) * HID_SZ + n0 + nn]      = acc0[r];
      icur_s[(r + mo) * HID_SZ + n0 + 16 + nn] = acc1[r];
    }
    __syncthreads();
#pragma unroll
    for (int r = 0; r < 16; r++) {
      float ic = icur_s[r * HID_SZ + tid];
      float ri = (imem[r] > 1.0f) ? 1.0f : 0.0f;      // reset from old mem
      imem[r] = BETA * imem[r] + ic - ri;
      isum[r] += (imem[r] > 1.0f) ? 1.0f : 0.0f;
    }
  }

  // pooled -> icur_s (each thread overwrites only its own slots: no race)
#pragma unroll
  for (int r = 0; r < 16; r++)
    icur_s[r * HID_SZ + tid] = isum[r] * (1.0f / T_SZ);
  __syncthreads();

  // out[16 x 8] = pooled @ W_out^T + b_out (tiny; plain VALU dot products)
  if (tid < 16 * OUT_SZ) {
    int r = tid >> 3, o = tid & 7;
    float acc = 0.f;
    for (int c = 0; c < HID_SZ; c++)
      acc += icur_s[r * HID_SZ + c] * W_out[o * HID_SZ + c];
    out[(row0 + r) * OUT_SZ + o] = acc + b_out[o];
  }
}

// ---------------------------------------------------------------------------
// Host launch
// ---------------------------------------------------------------------------

extern "C" void kernel_launch(void* const* d_in, const int* in_sizes, int n_in,
                              void* d_out, int out_size, void* d_ws, size_t ws_size,
                              hipStream_t stream) {
  const float* dist_lat = (const float*)d_in[0];
  const float* az_lat   = (const float*)d_in[1];
  const float* el_lat   = (const float*)d_in[2];
  const float* W_drive  = (const float*)d_in[3];
  const float* b_drive  = (const float*)d_in[4];
  const float* W_rec    = (const float*)d_in[5];
  const float* b_rec    = (const float*)d_in[6];
  const float* decayraw = (const float*)d_in[7];
  const float* freqraw  = (const float*)d_in[8];
  const float* W_fus    = (const float*)d_in[9];
  const float* b_fus    = (const float*)d_in[10];
  const float* W_int    = (const float*)d_in[11];
  const float* b_int    = (const float*)d_in[12];
  const float* W_out    = (const float*)d_in[13];
  const float* b_out    = (const float*)d_in[14];
  float* out = (float*)d_out;

  // Workspace layout (bytes)
  char* ws = (char*)d_ws;
  _Float16* Wd_h    = (_Float16*)(ws + 0);          //  64K f16  (128 KB)
  _Float16* Wr_h    = (_Float16*)(ws + 131072);     //  64K f16  (128 KB)
  _Float16* Wi_h    = (_Float16*)(ws + 262144);     // 256K f16  (512 KB)
  _Float16* Wf_h    = (_Float16*)(ws + 786432);     // 384K f16  (768 KB)
  _Float16* dl_h    = (_Float16*)(ws + 1572864);    // 512K f16  (1 MB)
  _Float16* fused_h = (_Float16*)(ws + 2621440);    // 2048x768 f16 (3 MB)
  float*    drive0  = (float*)   (ws + 5767168);    // 2048x256 f32 (2 MB)
  float*    fcur    = (float*)   (ws + 7864320);    // 2048x512 f32 (4 MB)
  float*    decay   = (float*)   (ws + 12058624);   // 256 f32
  float*    freq    = (float*)   (ws + 12059648);   // 256 f32

  // 1) weight / latent conversions to f16
  cvt_f32_to_f16<<<(H_SZ * H_SZ + 255) / 256, 256, 0, stream>>>(W_drive, Wd_h, H_SZ * H_SZ);
  cvt_f32_to_f16<<<(H_SZ * H_SZ + 255) / 256, 256, 0, stream>>>(W_rec, Wr_h, H_SZ * H_SZ);
  cvt_f32_to_f16<<<(HID_SZ * HID_SZ + 255) / 256, 256, 0, stream>>>(W_int, Wi_h, HID_SZ * HID_SZ);
  cvt_f32_to_f16<<<(HID_SZ * 3 * H_SZ + 255) / 256, 256, 0, stream>>>(W_fus, Wf_h, HID_SZ * 3 * H_SZ);
  cvt_f32_to_f16<<<(B_SZ * H_SZ + 255) / 256, 256, 0, stream>>>(dist_lat, dl_h, B_SZ * H_SZ);
  decay_freq_kernel<<<1, H_SZ, 0, stream>>>(decayraw, freqraw, decay, freq);
  pack_azel<<<(B_SZ * H_SZ + 255) / 256, 256, 0, stream>>>(az_lat, el_lat, fused_h);

  // 2) drive0 = dist_lat @ W_drive^T + b_drive   [2048,256]
  {
    int tiles = (B_SZ / 16) * (H_SZ / 16);          // 2048 tiles
    gemm_wmma<<<tiles / 8, 256, 0, stream>>>(dl_h, Wd_h, b_drive, drive0,
                                             B_SZ, H_SZ, H_SZ);
  }

  // 3) distance recurrent loop -> fused_h[:, 0:256]
  dist_loop<<<B_SZ / 16, 256, 0, stream>>>(drive0, decay, freq, Wr_h, b_rec, fused_h);

  // 4) fcur = fused @ W_fus^T + b_fus   [2048,512], K=768
  {
    int tiles = (B_SZ / 16) * (HID_SZ / 16);        // 4096 tiles
    gemm_wmma<<<tiles / 8, 256, 0, stream>>>(fused_h, Wf_h, b_fus, fcur,
                                             B_SZ, HID_SZ, 3 * H_SZ);
  }

  // 5) two-layer LIF loop + output projection -> d_out [2048,8]
  lif_loop<<<B_SZ / 16, 512, 0, stream>>>(fcur, Wi_h, b_int, W_out, b_out, out);
}
